// SGRUCell_28939489640763
// MI455X (gfx1250) — matrix operand order repaired and checked by reading, not compiled, not run on hardware
//
#include <hip/hip_runtime.h>
#include <hip/hip_bf16.h>

// ---- sizes ----
#define TT 16
#define BB 16
#define II 256
#define HH 512
#define MM 128
#define KTOP 32
#define G4H 2048   // 4*H

typedef __attribute__((ext_vector_type(16))) __bf16 v16bf;
typedef __attribute__((ext_vector_type(8)))  float  v8f;

union V16 { v16bf v; unsigned short u[16]; };

__device__ __forceinline__ unsigned short f2bf(float f) {
  unsigned int u = __float_as_uint(f);
  unsigned int r = (u + 0x7FFFu + ((u >> 16) & 1u)) >> 16;
  return (unsigned short)r;
}
__device__ __forceinline__ float sigf(float x) { return 1.0f / (1.0f + expf(-x)); }
__device__ __forceinline__ float clamp5(float x) { return fminf(fmaxf(x, -5.0f), 5.0f); }

// ---------------- f32 -> bf16 conversion ----------------
__global__ void cvt_bf16_kernel(const float* __restrict__ in, unsigned short* __restrict__ out, int n) {
  int i = blockIdx.x * blockDim.x + threadIdx.x;
  if (i < n) out[i] = f2bf(in[i]);
}

// ---------------- state init ----------------
__global__ void init_state_kernel(const float* __restrict__ h0, const float* __restrict__ v0,
                                  const float* __restrict__ te0,
                                  float* __restrict__ hb, float* __restrict__ vb,
                                  float* __restrict__ teb, unsigned short* __restrict__ hbf) {
  int i = blockIdx.x * blockDim.x + threadIdx.x;
  if (i < BB * HH) {
    float h = h0[i];
    hb[i] = h; vb[i] = v0[i]; teb[i] = te0[i]; hbf[i] = f2bf(h);
  }
}

// ---------------- generic WMMA GEMM: C[M,N] = A[M,K] * W[N,K]^T + bias ----------------
// A, W in bf16 (row-major), C f32.  One wave (32 threads) per 16x16 tile.
__global__ void wmma_gemm_bias(const unsigned short* __restrict__ A,
                               const unsigned short* __restrict__ W,
                               const float* __restrict__ bias,
                               float* __restrict__ C,
                               int Kdim, int ntilesN) {
  int tile = blockIdx.x;
  int tm = tile / ntilesN;
  int tn = tile % ntilesN;
  int lane = threadIdx.x & 31;
  int mrow = lane & 15;           // A row within tile / C col within tile
  int half = lane >> 4;
  int N = ntilesN * 16;

  const unsigned short* Arow = A + (size_t)(tm * 16 + mrow) * Kdim;
  const unsigned short* Wrow = W + (size_t)(tn * 16 + mrow) * Kdim; // column n = lane&15

  v8f acc = {};
  int nk = Kdim >> 5;
  for (int kt = 0; kt < nk; ++kt) {
    int k0 = kt << 5;
    V16 a, b;
#pragma unroll
    for (int p = 0; p < 8; ++p) {
      int kb = k0 + ((p >> 2) << 4) + (half << 3) + ((p & 3) << 1);
      a.u[2 * p]     = Arow[kb];
      a.u[2 * p + 1] = Arow[kb + 1];
    }
    const unsigned short* bp = Wrow + k0 + (half << 3);
#pragma unroll
    for (int e = 0; e < 16; ++e) b.u[e] = bp[(e & 7) + ((e >> 3) << 4)];
    acc = __builtin_amdgcn_wmma_f32_16x16x32_bf16(false, a.v, false, b.v,
                                                  (short)0, acc, false, false);
  }
  int col = tn * 16 + (lane & 15);
  float bv = bias[col];
#pragma unroll
  for (int r = 0; r < 8; ++r) {
    int row = tm * 16 + r + (half << 3);
    C[(size_t)row * N + col] = acc[r] + bv;
  }
}

// ---------------- row LayerNorm (+ optional addend), in-place safe ----------------
__global__ void ln_rows_kernel(const float* __restrict__ in, const float* __restrict__ gamma,
                               const float* __restrict__ beta, const float* __restrict__ addend,
                               float* __restrict__ out, int ncols) {
  __shared__ float s1[256];
  __shared__ float s2[256];
  int row = blockIdx.x, tid = threadIdx.x, nthr = blockDim.x;
  const float* r = in + (size_t)row * ncols;
  float s = 0.f, sq = 0.f;
  for (int c = tid; c < ncols; c += nthr) { float y = r[c]; s += y; sq += y * y; }
  s1[tid] = s; s2[tid] = sq; __syncthreads();
  for (int off = 128; off > 0; off >>= 1) {
    if (tid < off) { s1[tid] += s1[tid + off]; s2[tid] += s2[tid + off]; }
    __syncthreads();
  }
  float mu = s1[0] / (float)ncols;
  float var = s2[0] / (float)ncols - mu * mu;
  float rstd = rsqrtf(var + 1e-5f);
  const float* ad = addend ? addend + (size_t)row * ncols : nullptr;
  for (int c = tid; c < ncols; c += nthr) {
    float y = (r[c] - mu) * rstd * gamma[c] + beta[c];
    if (ad) y += ad[c];
    out[(size_t)row * ncols + c] = y;
  }
}

// ---------------- initial dU . h dot (step 0 only) ----------------
__global__ void du_h_dot_kernel(const float* __restrict__ dU, const float* __restrict__ h,
                                float* __restrict__ dot) {
  int row = blockIdx.x;              // b*H + i
  int b = row >> 9;
  int tid = threadIdx.x;             // 128 threads, float4 over j
  const float4* D = (const float4*)(dU + (size_t)row * HH);
  const float4* Hv = (const float4*)(h + (size_t)b * HH);
  float4 d4 = D[tid], h4 = Hv[tid];
  float p = d4.x * h4.x + d4.y * h4.y + d4.z * h4.z + d4.w * h4.w;
  __shared__ float red[128];
  red[tid] = p; __syncthreads();
  for (int off = 64; off > 0; off >>= 1) {
    if (tid < off) red[tid] += red[tid + off];
    __syncthreads();
  }
  if (tid == 0) dot[row] = red[0];
}

// ---------------- gates + v/h/te update (dot pre-computed) ----------------
__global__ void gates_state_kernel(const float* __restrict__ g, const float* __restrict__ dot,
                                   float* __restrict__ v, const float* __restrict__ te_old,
                                   float* __restrict__ h_new, float* __restrict__ te_new,
                                   unsigned short* __restrict__ hbf,
                                   const float* __restrict__ alpha, const float* __restrict__ EI,
                                   float* __restrict__ seq_t) {
  int row = blockIdx.x * blockDim.x + threadIdx.x;
  if (row >= BB * HH) return;
  int b = row >> 9, i = row & (HH - 1);
  const float* gr = g + (size_t)b * G4H;
  float pg = EI[i] * fabsf(alpha[i]);
  float dv = gr[3 * HH + i] + pg * dot[row];
  float fg = sigf(gr[i]);
  float ig = sigf(gr[HH + i]);
  float vn = fg * v[row] + ig * dv;
  v[row] = vn;
  float hn = tanhf(vn);
  h_new[row] = hn;
  hbf[row] = f2bf(hn);
  seq_t[row] = hn;
  te_new[row] = fg * te_old[row] + ig * hn;
}

// ---------------- modulatory pathway: h2mod GEMV, kWTA(top-32 of 128), mod2h GEMV ----------------
__global__ void modpath_kernel(const float* __restrict__ h_new,
                               const float* __restrict__ h2mod_w, const float* __restrict__ h2mod_b,
                               const float* __restrict__ mod2h_w, const float* __restrict__ mod2h_b,
                               float* __restrict__ m2h) {
  int b = blockIdx.x;
  int m = threadIdx.x;               // 128 threads
  __shared__ float hs[HH];
  __shared__ float vals[MM];
  __shared__ float thresh;
  for (int idx = m; idx < HH; idx += MM) hs[idx] = h_new[(size_t)b * HH + idx];
  __syncthreads();
  float acc = h2mod_b[m];
  const float* wrow = h2mod_w + (size_t)m * HH;
  for (int k = 0; k < HH; ++k) acc += hs[k] * wrow[k];
  vals[m] = acc;
  __syncthreads();
  // stable rank: unique permutation of 0..127; rank==K-1 -> K-th largest value
  float vm = vals[m];
  int rank = 0;
  for (int j = 0; j < MM; ++j) {
    float vj = vals[j];
    rank += (vj > vm) || (vj == vm && j < m);
  }
  if (rank == KTOP - 1) thresh = vm;
  __syncthreads();
  float w = (vm >= thresh) ? vm : 0.0f;
  __syncthreads();
  vals[m] = w;
  __syncthreads();
  for (int r = 0; r < HH / MM; ++r) {
    int i = m + r * MM;
    float a2 = mod2h_b[i];
    const float* mw = mod2h_w + (size_t)i * MM;
    for (int k = 0; k < MM; ++k) a2 += vals[k] * mw[k];
    m2h[(size_t)b * HH + i] = a2;
  }
}

// ---------------- TE / dU streaming update, fused with next-step dU.h dot ----------------
__global__ void te_du_update_kernel(const float* __restrict__ g, const float* __restrict__ tau_U,
                                    const float* __restrict__ m2h,
                                    const float* __restrict__ h_old, const float* __restrict__ te_old,
                                    const float* __restrict__ h_new, const float* __restrict__ te_new,
                                    float* __restrict__ TE, float* __restrict__ dU,
                                    float* __restrict__ dot_out) {
  int row = blockIdx.x;              // b*H + i
  int b = row >> 9, i = row & (HH - 1);
  int tid = threadIdx.x;             // 128 threads, float4 over j
  float s   = sigf(g[(size_t)b * G4H + 2 * HH + i]);
  float os  = 1.0f - s;
  float dec = sigf(tau_U[i]);
  float hni = h_new[row], tni = te_new[row], mi = m2h[row];
  const float4* teo = (const float4*)(te_old + (size_t)b * HH);
  const float4* ho  = (const float4*)(h_old + (size_t)b * HH);
  const float4* hn4 = (const float4*)(h_new + (size_t)b * HH);
  float4* TEp = (float4*)(TE + (size_t)row * HH);
  float4* dUp = (float4*)(dU + (size_t)row * HH);
  float4 t4 = teo[tid], h4 = ho[tid], hnv = hn4[tid];
  float4 T4 = TEp[tid], D4 = dUp[tid];
  float4 Tn, Dn;
  Tn.x = os * T4.x + s * (hni * t4.x - tni * h4.x);
  Tn.y = os * T4.y + s * (hni * t4.y - tni * h4.y);
  Tn.z = os * T4.z + s * (hni * t4.z - tni * h4.z);
  Tn.w = os * T4.w + s * (hni * t4.w - tni * h4.w);
  Dn.x = clamp5(dec * D4.x + mi * Tn.x);
  Dn.y = clamp5(dec * D4.y + mi * Tn.y);
  Dn.z = clamp5(dec * D4.z + mi * Tn.z);
  Dn.w = clamp5(dec * D4.w + mi * Tn.w);
  TEp[tid] = Tn;
  dUp[tid] = Dn;
  // fused next-step dot: dU_new[row,:] . h_new[b,:]
  float p = Dn.x * hnv.x + Dn.y * hnv.y + Dn.z * hnv.z + Dn.w * hnv.w;
  __shared__ float red[128];
  red[tid] = p; __syncthreads();
  for (int off = 64; off > 0; off >>= 1) {
    if (tid < off) red[tid] += red[tid + off];
    __syncthreads();
  }
  if (tid == 0) dot_out[row] = red[0];
}

// ---------------- host launcher ----------------
extern "C" void kernel_launch(void* const* d_in, const int* in_sizes, int n_in,
                              void* d_out, int out_size, void* d_ws, size_t ws_size,
                              hipStream_t stream) {
  (void)in_sizes; (void)n_in; (void)out_size; (void)ws_size;
  const float* x       = (const float*)d_in[0];   // (T,B,I)
  const float* h0      = (const float*)d_in[1];
  const float* v0      = (const float*)d_in[2];
  const float* dU0     = (const float*)d_in[3];
  const float* te0     = (const float*)d_in[4];
  const float* TE0     = (const float*)d_in[5];
  const float* x2h_w   = (const float*)d_in[6];
  const float* x2h_b   = (const float*)d_in[7];
  const float* h2h_w   = (const float*)d_in[8];
  const float* h2h_b   = (const float*)d_in[9];
  const float* lnx_g   = (const float*)d_in[10];
  const float* lnx_b   = (const float*)d_in[11];
  const float* lnh_g   = (const float*)d_in[12];
  const float* lnh_b   = (const float*)d_in[13];
  const float* alpha   = (const float*)d_in[14];
  const float* tau_U   = (const float*)d_in[15];
  const float* h2mod_w = (const float*)d_in[16];
  const float* h2mod_b = (const float*)d_in[17];
  const float* mod2h_w = (const float*)d_in[18];
  const float* mod2h_b = (const float*)d_in[19];
  const float* EI      = (const float*)d_in[20];

  float* out = (float*)d_out;
  float* o_v   = out;
  float* o_h   = out + 8192;
  float* o_dU  = out + 16384;                 // 16*512*512
  float* o_te  = out + 16384 + 4194304;
  float* o_TE  = out + 16384 + 4194304 + 8192;
  float* o_seq = out + 16384 + 4194304 + 8192 + 4194304;

  // workspace layout
  char* w = (char*)d_ws;
  size_t off = 0;
  auto take = [&](size_t bytes) { char* p = w + off; off = (off + bytes + 255) & ~(size_t)255; return p; };
  unsigned short* x_bf    = (unsigned short*)take((size_t)TT * BB * II * 2);
  unsigned short* x2h_bf  = (unsigned short*)take((size_t)G4H * II * 2);
  unsigned short* h2h_bf  = (unsigned short*)take((size_t)G4H * HH * 2);
  float* Wx     = (float*)take((size_t)TT * BB * G4H * 4);   // raw -> LN'ed in place
  float* Wh     = (float*)take((size_t)BB * G4H * 4);
  float* gbuf   = (float*)take((size_t)BB * G4H * 4);
  float* vbuf   = (float*)take((size_t)BB * HH * 4);
  float* hbuf0  = (float*)take((size_t)BB * HH * 4);
  float* hbuf1  = (float*)take((size_t)BB * HH * 4);
  float* tebuf0 = (float*)take((size_t)BB * HH * 4);
  float* tebuf1 = (float*)take((size_t)BB * HH * 4);
  unsigned short* hbf = (unsigned short*)take((size_t)BB * HH * 2);
  float* m2h    = (float*)take((size_t)BB * HH * 4);
  float* dotbuf = (float*)take((size_t)BB * HH * 4);

  // state init: dU/TE live directly in d_out (L2-resident working set)
  hipMemcpyAsync(o_dU, dU0, (size_t)BB * HH * HH * 4, hipMemcpyDeviceToDevice, stream);
  hipMemcpyAsync(o_TE, TE0, (size_t)BB * HH * HH * 4, hipMemcpyDeviceToDevice, stream);
  init_state_kernel<<<dim3(32), dim3(256), 0, stream>>>(h0, v0, te0, hbuf0, vbuf, tebuf0, hbf);

  // bf16 conversions
  {
    int n1 = TT * BB * II, n2 = G4H * II, n3 = G4H * HH;
    cvt_bf16_kernel<<<dim3((n1 + 255) / 256), dim3(256), 0, stream>>>(x, x_bf, n1);
    cvt_bf16_kernel<<<dim3((n2 + 255) / 256), dim3(256), 0, stream>>>(x2h_w, x2h_bf, n2);
    cvt_bf16_kernel<<<dim3((n3 + 255) / 256), dim3(256), 0, stream>>>(h2h_w, h2h_bf, n3);
  }

  // Wx for ALL timesteps in one WMMA GEMM: (256 x 256) @ (256 x 2048), then LN in place
  wmma_gemm_bias<<<dim3(16 * 128), dim3(32), 0, stream>>>(x_bf, x2h_bf, x2h_b, Wx, II, 128);
  ln_rows_kernel<<<dim3(TT * BB), dim3(256), 0, stream>>>(Wx, lnx_g, lnx_b, nullptr, Wx, G4H);

  // initial dU0 . h0 dot
  du_h_dot_kernel<<<dim3(BB * HH), dim3(128), 0, stream>>>(o_dU, hbuf0, dotbuf);

  for (int t = 0; t < TT; ++t) {
    float* h_old  = (t & 1) ? hbuf1  : hbuf0;
    float* h_newb = (t & 1) ? hbuf0  : hbuf1;
    float* te_old = (t & 1) ? tebuf1 : tebuf0;
    float* te_newb= (t & 1) ? tebuf0 : tebuf1;

    // Wh = h @ h2h_w^T + b   (WMMA, 1 x 128 tiles, K=512)
    wmma_gemm_bias<<<dim3(128), dim3(32), 0, stream>>>(hbf, h2h_bf, h2h_b, Wh, HH, 128);
    // g = LN(Wh) + WxLN[t]
    ln_rows_kernel<<<dim3(BB), dim3(256), 0, stream>>>(Wh, lnh_g, lnh_b,
                                                       Wx + (size_t)t * BB * G4H, gbuf, G4H);
    // gates + v/h/te update (dot pre-fused from previous step)
    gates_state_kernel<<<dim3(32), dim3(256), 0, stream>>>(gbuf, dotbuf, vbuf, te_old,
                                                           h_newb, te_newb, hbf, alpha, EI,
                                                           o_seq + (size_t)t * BB * HH);
    // mod pathway: h2mod -> kWTA top-32 -> mod2h
    modpath_kernel<<<dim3(BB), dim3(128), 0, stream>>>(h_newb, h2mod_w, h2mod_b,
                                                       mod2h_w, mod2h_b, m2h);
    // TE / dU streaming update + fused next-step dU.h dot
    te_du_update_kernel<<<dim3(BB * HH), dim3(128), 0, stream>>>(gbuf, tau_U, m2h,
                                                                 h_old, te_old, h_newb, te_newb,
                                                                 o_TE, o_dU, dotbuf);
  }

  // final small states (dU/TE already in place in d_out). After t=15 (odd): new state in buf0.
  hipMemcpyAsync(o_v,  vbuf,   (size_t)BB * HH * 4, hipMemcpyDeviceToDevice, stream);
  hipMemcpyAsync(o_h,  hbuf0,  (size_t)BB * HH * 4, hipMemcpyDeviceToDevice, stream);
  hipMemcpyAsync(o_te, tebuf0, (size_t)BB * HH * 4, hipMemcpyDeviceToDevice, stream);
}
